// Graph_module_net_0_18631568130103
// MI455X (gfx1250) — matrix operand order repaired and checked by reading, not compiled, not run on hardware
//
#include <hip/hip_runtime.h>

// ---------------------------------------------------------------------------
// Graph attention block for MI455X (gfx1250, wave32, WMMA).
//
// All heavy math runs through v_wmma_f32_16x16x32_bf16 (bf16 in, f32 accum).
// The reference's top_k-union mask is provably all-ones (diagonal cosine == 1
// is always in each row's top-4), so it is folded away.
//
// GEMM kernels use a 16x64 per-wave tile (4 accumulators): the A fragment is
// loaded once per 32-wide k-chunk and reused by 4 WMMAs, quadrupling
// register-level reuse on the data-movement-limited path.
// ---------------------------------------------------------------------------

typedef __attribute__((ext_vector_type(16))) __bf16 v16bf;
typedef __attribute__((ext_vector_type(8)))  __bf16 v8bf;
typedef __attribute__((ext_vector_type(8)))  float  v8f;
typedef __attribute__((ext_vector_type(4)))  float  v4f;

static constexpr int Bb = 4;      // batch
static constexpr int Nn = 2048;   // nodes
static constexpr int Dd = 256;    // C == M == O
static constexpr int Gg = 4;      // conv groups (64 ch per group)

// ---- WMMA fragment loaders (layouts per CDNA5 ISA 7.12.2) ------------------
// A (16x32 MxK, bf16): lane l -> row m = l%16; lane-half g = l/16;
//   halves 0..7  hold K = kc + 8g .. +8
//   halves 8..15 hold K = kc + 16 + 8g .. +8
__device__ inline v16bf frag_a_bf(const __bf16* __restrict__ row, int kc, int lg) {
  v8bf lo = *(const v8bf*)(row + kc + 8 * lg);
  v8bf hi = *(const v8bf*)(row + kc + 16 + 8 * lg);
  v16bf r;
#pragma unroll
  for (int i = 0; i < 8; ++i) { r[i] = lo[i]; r[i + 8] = hi[i]; }
  return r;
}

// A fragment from an f32 row-major matrix (weights), converted to bf16.
__device__ inline v16bf frag_a_f32(const float* __restrict__ row, int kc, int lg) {
  v16bf r;
#pragma unroll
  for (int i = 0; i < 8; ++i) {
    r[i]     = (__bf16)row[kc + 8 * lg + i];
    r[i + 8] = (__bf16)row[kc + 16 + 8 * lg + i];
  }
  return r;
}

// B (32x16 KxN, bf16): lane l -> col n = l%16; halves 0..7 -> K = kc + 8*(l/16) .. +8, halves 8..15 -> K = kc + 16 + 8*(l/16) .. +8.
// `row` points at the K-contiguous vector belonging to this lane's column.
__device__ inline v16bf frag_b_bf(const __bf16* __restrict__ row, int kc, int lg) {
  v8bf lo = *(const v8bf*)(row + kc + 8 * lg);
  v8bf hi = *(const v8bf*)(row + kc + 16 + 8 * lg);
  v16bf r;
#pragma unroll
  for (int i = 0; i < 8; ++i) { r[i] = lo[i]; r[i + 8] = hi[i]; }
  return r;
}

__device__ inline v8f wmma_bf16(v16bf a, v16bf b, v8f c) {
  return __builtin_amdgcn_wmma_f32_16x16x32_bf16(false, a, false, b, (short)0, c,
                                                 false, false);
}

// ---- prep: x f32 -> x_bf + xn_bf (one wave per node row of 256) ------------
__global__ __launch_bounds__(256) void prep_x_kernel(
    const float* __restrict__ x, __bf16* __restrict__ x_bf,
    __bf16* __restrict__ xn_bf) {
  int wave = blockIdx.x * 8 + (threadIdx.x >> 5);
  int lane = threadIdx.x & 31;
  const float* row = x + (size_t)wave * Dd + lane * 8;
  v4f a = *(const v4f*)row;
  v4f b = *(const v4f*)(row + 4);
  float ss = 0.f;
#pragma unroll
  for (int i = 0; i < 4; ++i) ss += a[i] * a[i] + b[i] * b[i];
#pragma unroll
  for (int m = 16; m >= 1; m >>= 1) ss += __shfl_xor(ss, m, 32);
  float nrm = sqrtf(ss);
  float inv = 1.f / (nrm > 1e-8f ? nrm : 1e-8f);
  v8bf xo, no;
#pragma unroll
  for (int i = 0; i < 4; ++i) {
    xo[i] = (__bf16)a[i];         xo[i + 4] = (__bf16)b[i];
    no[i] = (__bf16)(a[i] * inv); no[i + 4] = (__bf16)(b[i] * inv);
  }
  *(v8bf*)(x_bf + (size_t)wave * Dd + lane * 8) = xo;
  *(v8bf*)(xn_bf + (size_t)wave * Dd + lane * 8) = no;
}

// ---- normalize a bf16 [B*N, 256] matrix row-wise ---------------------------
__global__ __launch_bounds__(256) void normalize_kernel(
    const __bf16* __restrict__ in, __bf16* __restrict__ out) {
  int wave = blockIdx.x * 8 + (threadIdx.x >> 5);
  int lane = threadIdx.x & 31;
  v8bf v = *(const v8bf*)(in + (size_t)wave * Dd + lane * 8);
  float f[8], ss = 0.f;
#pragma unroll
  for (int i = 0; i < 8; ++i) { f[i] = (float)v[i]; ss += f[i] * f[i]; }
#pragma unroll
  for (int m = 16; m >= 1; m >>= 1) ss += __shfl_xor(ss, m, 32);
  float nrm = sqrtf(ss);
  float inv = 1.f / (nrm > 1e-8f ? nrm : 1e-8f);
  v8bf o;
#pragma unroll
  for (int i = 0; i < 8; ++i) o[i] = (__bf16)(f[i] * inv);
  *(v8bf*)(out + (size_t)wave * Dd + lane * 8) = o;
}

// ---- gram + masks: At[b,r,c] = (relu(v_r.v_c)*roi + diag)/4 ----------------
// At is the TRANSPOSE of the reference's A (At[k,n] = A[n,k]) so the later
// "X @ A" GEMM reads it K-contiguously as the B operand.
// Wave tile: 16 rows x 64 cols (4 accumulators, A fragment reused 4x).
__global__ __launch_bounds__(256) void gram_mask_kernel(
    const __bf16* __restrict__ V,          // [B*N, 256] normalized
    const float* __restrict__ masks_roi,   // [B,N,N]
    const float* __restrict__ score_mask,  // [B,N]
    __bf16* __restrict__ At) {             // [B,N,N]
  int wave = blockIdx.x * 8 + (threadIdx.x >> 5);
  int lane = threadIdx.x & 31;
  int lg = lane >> 4, li = lane & 15;
  constexpr int TPB = (Nn / 16) * (Nn / 64);  // 128 * 32
  int b = wave / TPB, t = wave % TPB;
  int r0 = (t / (Nn / 64)) * 16, c0 = (t % (Nn / 64)) * 64;
  const __bf16* arow = V + (size_t)(b * Nn + r0 + li) * Dd;
  const __bf16* brow0 = V + (size_t)(b * Nn + c0 + li) * Dd;
  v8f acc[4];
#pragma unroll
  for (int j = 0; j < 4; ++j) acc[j] = {0.f, 0.f, 0.f, 0.f, 0.f, 0.f, 0.f, 0.f};
#pragma unroll
  for (int kc = 0; kc < Dd; kc += 32) {
    v16bf a = frag_a_bf(arow, kc, lg);
#pragma unroll
    for (int j = 0; j < 4; ++j)
      acc[j] = wmma_bf16(a, frag_b_bf(brow0 + (size_t)(16 * j) * Dd, kc, lg), acc[j]);
  }
#pragma unroll
  for (int j = 0; j < 4; ++j) {
    int col = c0 + 16 * j + li;
    float smc = score_mask[b * Nn + col];
#pragma unroll
    for (int r = 0; r < 8; ++r) {
      int row = r0 + r + 8 * lg;
      float tt = acc[j][r] > 0.f ? acc[j][r] : 0.f;
      float v = tt * masks_roi[((size_t)b * Nn + row) * Nn + col] * smc;
      if (row == col && score_mask[b * Nn + row] == 0.f) v += 1.f;
      At[((size_t)b * Nn + row) * Nn + col] = (__bf16)(v * 0.25f);
    }
  }
}

// ---- grouped 1x1 conv + bias + relu: out[b,m,n] (feat-major) ---------------
// Wave tile: 16 out-channels x 64 nodes (weight fragment reused 4x).
__global__ __launch_bounds__(256) void gconv_kernel(
    const float* __restrict__ W,        // [G, 64, 64]
    const float* __restrict__ bias,     // [256]
    const __bf16* __restrict__ inNM,    // [B, N, 256] node-major
    __bf16* __restrict__ out) {         // [B, 256, N]
  int wave = blockIdx.x * 8 + (threadIdx.x >> 5);
  int lane = threadIdx.x & 31;
  int lg = lane >> 4, li = lane & 15;
  constexpr int TPB = (Dd / 16) * (Nn / 64);  // 16 * 32
  int b = wave / TPB, t = wave % TPB;
  int m0 = (t / (Nn / 64)) * 16, n0 = (t % (Nn / 64)) * 64;
  int g = m0 >> 6;
  const float* wrow = W + g * 64 * 64 + ((m0 & 63) + li) * 64;
  const __bf16* brow0 = inNM + (size_t)(b * Nn + n0 + li) * Dd + g * 64;
  v8f acc[4];
#pragma unroll
  for (int j = 0; j < 4; ++j) acc[j] = {0.f, 0.f, 0.f, 0.f, 0.f, 0.f, 0.f, 0.f};
#pragma unroll
  for (int kc = 0; kc < 64; kc += 32) {
    v16bf a = frag_a_f32(wrow, kc, lg);
#pragma unroll
    for (int j = 0; j < 4; ++j)
      acc[j] = wmma_bf16(a, frag_b_bf(brow0 + (size_t)(16 * j) * Dd, kc, lg), acc[j]);
  }
#pragma unroll
  for (int j = 0; j < 4; ++j) {
    int col = n0 + 16 * j + li;
#pragma unroll
    for (int r = 0; r < 8; ++r) {
      int row = m0 + r + 8 * lg;
      float v = acc[j][r] + bias[row];
      out[((size_t)b * Dd + row) * Nn + col] = (__bf16)(v > 0.f ? v : 0.f);
    }
  }
}

// ---- apply attention: R[m,k] = sum_n L[m,n] * At[k,n] ----------------------
// Wave tile: 16 features x 64 nodes (A fragment reused 4x), K = N = 2048.
// MODE 0: R += L (residual), store bf16 feat-major AND bf16 node-major.
// MODE 1: store f32 feat-major only.
template <int MODE>
__global__ __launch_bounds__(256) void apply_attn_kernel(
    const __bf16* __restrict__ L,    // [B, 256, N]
    const __bf16* __restrict__ At,   // [B, N, N]
    __bf16* __restrict__ outF,       // [B, 256, N]   (MODE 0)
    __bf16* __restrict__ outT,       // [B, N, 256]   (MODE 0)
    float* __restrict__ outF32) {    // [B, 256, N]   (MODE 1)
  int wave = blockIdx.x * 8 + (threadIdx.x >> 5);
  int lane = threadIdx.x & 31;
  int lg = lane >> 4, li = lane & 15;
  constexpr int TPB = (Dd / 16) * (Nn / 64);  // 16 * 32
  int b = wave / TPB, t = wave % TPB;
  int m0 = (t / (Nn / 64)) * 16, k0 = (t % (Nn / 64)) * 64;
  const __bf16* arow = L + (size_t)(b * Dd + m0 + li) * Nn;
  const __bf16* brow0 = At + (size_t)(b * Nn + k0 + li) * Nn;
  v8f acc[4];
#pragma unroll
  for (int j = 0; j < 4; ++j) acc[j] = {0.f, 0.f, 0.f, 0.f, 0.f, 0.f, 0.f, 0.f};
#pragma unroll 4
  for (int kc = 0; kc < Nn; kc += 32) {
    v16bf a = frag_a_bf(arow, kc, lg);
#pragma unroll
    for (int j = 0; j < 4; ++j)
      acc[j] = wmma_bf16(a, frag_b_bf(brow0 + (size_t)(16 * j) * Nn, kc, lg), acc[j]);
  }
#pragma unroll
  for (int j = 0; j < 4; ++j) {
    int col = k0 + 16 * j + li;
#pragma unroll
    for (int r = 0; r < 8; ++r) {
      int row = m0 + r + 8 * lg;
      if (MODE == 0) {
        float v = acc[j][r] + (float)L[((size_t)b * Dd + row) * Nn + col];
        __bf16 bv = (__bf16)v;
        outF[((size_t)b * Dd + row) * Nn + col] = bv;
        outT[((size_t)b * Nn + col) * Dd + row] = bv;
      } else {
        outF32[((size_t)b * Dd + row) * Nn + col] = acc[j][r];
      }
    }
  }
}

// ---- LayerNorm over O + residual + transpose to [B,N,O] f32 ----------------
__global__ __launch_bounds__(256) void ln_out_kernel(
    const float* __restrict__ m,     // [B, 256, N]
    const __bf16* __restrict__ o2,   // [B, 256, N]
    const float* __restrict__ lnw, const float* __restrict__ lnb,
    float* __restrict__ out) {       // [B, N, 256]
  int idx = blockIdx.x * 256 + threadIdx.x;  // node id over B*N
  int b = idx / Nn, n = idx % Nn;
  float s = 0.f, s2 = 0.f;
  for (int o = 0; o < Dd; ++o) {
    float v = m[((size_t)b * Dd + o) * Nn + n];  // coalesced across lanes (n)
    s += v; s2 += v * v;
  }
  float mu = s * (1.f / Dd);
  float var = s2 * (1.f / Dd) - mu * mu;
  float inv = rsqrtf(var + 1e-6f);
  for (int o = 0; o < Dd; ++o) {
    float v = m[((size_t)b * Dd + o) * Nn + n];
    float ln = (v - mu) * inv * lnw[o] + lnb[o];
    out[((size_t)b * Nn + n) * Dd + o] =
        (float)o2[((size_t)b * Dd + o) * Nn + n] + ln;
  }
}

// ---------------------------------------------------------------------------
extern "C" void kernel_launch(void* const* d_in, const int* in_sizes, int n_in,
                              void* d_out, int out_size, void* d_ws, size_t ws_size,
                              hipStream_t stream) {
  const float* x          = (const float*)d_in[0];
  const float* masks_roi  = (const float*)d_in[1];
  const float* score_mask = (const float*)d_in[2];
  const float* w1         = (const float*)d_in[3];
  const float* b1         = (const float*)d_in[4];
  const float* w2         = (const float*)d_in[5];
  const float* b2         = (const float*)d_in[6];
  const float* ln_w       = (const float*)d_in[7];
  const float* ln_b       = (const float*)d_in[8];
  float* out = (float*)d_out;

  char* ws = (char*)d_ws;
  size_t off = 0;
  auto alloc = [&](size_t bytes) { char* p = ws + off; off += (bytes + 255) & ~(size_t)255; return p; };
  const size_t BNC = (size_t)Bb * Nn * Dd;   // 2,097,152 elems
  const size_t BNN = (size_t)Bb * Nn * Nn;   // 16,777,216 elems
  __bf16* x_bf   = (__bf16*)alloc(BNC * 2);
  __bf16* xn_bf  = (__bf16*)alloc(BNC * 2);  // reused as yn after first gram
  __bf16* At     = (__bf16*)alloc(BNN * 2);  // A1t then A2t
  __bf16* o1_bf  = (__bf16*)alloc(BNC * 2);  // [B,M,N]
  __bf16* o1b_bf = (__bf16*)alloc(BNC * 2);  // [B,M,N]
  __bf16* o1bT   = (__bf16*)alloc(BNC * 2);  // [B,N,M]
  __bf16* o2_bf  = (__bf16*)alloc(BNC * 2);  // [B,O,N]
  float*  m_f32  = (float*)alloc(BNC * 4);   // [B,O,N]

  const int rowBlocks  = (Bb * Nn) / 8;                       // 1024
  const int gramBlocks = (Bb * (Nn / 16) * (Nn / 64)) / 8;    // 2048
  const int gemmBlocks = (Bb * (Dd / 16) * (Nn / 64)) / 8;    // 256

  // 1) x -> bf16 copy + row-normalized copy
  prep_x_kernel<<<rowBlocks, 256, 0, stream>>>(x, x_bf, xn_bf);
  // 2) A1t = masked gram of xn
  gram_mask_kernel<<<gramBlocks, 256, 0, stream>>>(xn_bf, masks_roi, score_mask, At);
  // 3) o1 = gconv1(x)
  gconv_kernel<<<gemmBlocks, 256, 0, stream>>>(w1, b1, x_bf, o1_bf);
  // 4) o1b = o1 + o1 @ A1  (also emit node-major transpose)
  apply_attn_kernel<0><<<gemmBlocks, 256, 0, stream>>>(o1_bf, At, o1b_bf, o1bT, nullptr);
  // 5) yn = row-normalize(o1bT)   (reuses xn buffer)
  normalize_kernel<<<rowBlocks, 256, 0, stream>>>(o1bT, xn_bf);
  // 6) A2t = masked gram of yn    (reuses At buffer)
  gram_mask_kernel<<<gramBlocks, 256, 0, stream>>>(xn_bf, masks_roi, score_mask, At);
  // 7) o2 = gconv2(o1b)
  gconv_kernel<<<gemmBlocks, 256, 0, stream>>>(w2, b2, o1bT, o2_bf);
  // 8) m = o2 @ A2 (f32)
  apply_attn_kernel<1><<<gemmBlocks, 256, 0, stream>>>(o2_bf, At, nullptr, nullptr, m_f32);
  // 9) out = (o2 + LayerNorm(m^T)^T)^T
  ln_out_kernel<<<(Bb * Nn) / 256, 256, 0, stream>>>(m_f32, o2_bf, ln_w, ln_b, out);
}
